// SelectiveSSM2_52793738002854
// MI455X (gfx1250) — matrix operand order repaired and checked by reading, not compiled, not run on hardware
//
#include <hip/hip_runtime.h>
#include <hip/hip_bf16.h>

// ---------------------------------------------------------------- types
typedef __attribute__((ext_vector_type(16))) __bf16 v16bf;
typedef __attribute__((ext_vector_type(8)))  float  v8f;

union FragBF {
    v16bf v;
    uint4 q[2];
    unsigned short s[16];
};

#define B_SZ 2
#define SEQ_L 2048
#define DIM 2048
#define NS 128
#define RS 128
#define XPLD 384   // R + 2N
#define MTOK 4096  // B_SZ * SEQ_L

// ---------------------------------------------------------------- helpers
__device__ __forceinline__ unsigned short f2bf(float x) {
    union { float f; unsigned u; } v; v.f = x;
    unsigned r = v.u + 0x7FFFu + ((v.u >> 16) & 1u);   // round-to-nearest-even
    return (unsigned short)(r >> 16);
}
__device__ __forceinline__ float bf2f(unsigned short b) {
    union { unsigned u; float f; } v; v.u = ((unsigned)b) << 16;
    return v.f;
}

// ---------------------------------------------------------------- WMMA fragment loaders
// A-matrix 16x32 bf16 (M x K): lane&15 = M row; lane>>4 selects K sub-ranges.
// v16bf elems [0..7]  = K k0 + 8*half + 0..7
//             [8..15] = K k0 + 16 + 8*half + 0..7
__device__ __forceinline__ v16bf load_a_bf16(const unsigned short* A, int lda,
                                             int m0, int k0, int lane) {
    const int row = m0 + (lane & 15);
    const int h   = (lane >> 4) * 8;
    const unsigned short* p0 = A + (size_t)row * lda + k0 + h;
    FragBF f;
    f.q[0] = *(const uint4*)(p0);
    f.q[1] = *(const uint4*)(p0 + 16);
    return f.v;
}
__device__ __forceinline__ v16bf load_a_f32(const float* A, int lda,
                                            int m0, int k0, int lane) {
    const int row = m0 + (lane & 15);
    const int h   = (lane >> 4) * 8;
    const float* p0 = A + (size_t)row * lda + k0 + h;
    const float* p1 = p0 + 16;
    FragBF f;
#pragma unroll
    for (int i = 0; i < 8; ++i) {
        f.s[i]     = f2bf(p0[i]);
        f.s[8 + i] = f2bf(p1[i]);
    }
    return f.v;
}
// B-matrix 32x16 bf16 (K x N) = W^T with W row-major (N x K):
// lane&15 = N col; elems [0..7] = K 8*half + 0..7, [8..15] = K 16 + 8*half + 0..7.
__device__ __forceinline__ v16bf load_b(const unsigned short* W, int ldw,
                                        int n0, int k0, int lane) {
    const int col = n0 + (lane & 15);
    const int h   = (lane >> 4) * 8;
    const unsigned short* p = W + (size_t)col * ldw + k0 + h;
    FragBF f;
    f.q[0] = *(const uint4*)(p);
    f.q[1] = *(const uint4*)(p + 16);
    return f.v;
}

// ---------------------------------------------------------------- epilogues
struct EpiSelect {   // mask = sigmoid((acc+b)*gate); xg = xn * mask  (bf16 out)
    const float* sb; const float* sg;
    const unsigned short* xn; unsigned short* xg;
    __device__ void operator()(int m, int n, float acc) const {
        float l    = (acc + sb[n]) * sg[n];
        float mask = 1.0f / (1.0f + __expf(-l));
        size_t i   = (size_t)m * DIM + n;
        xg[i] = f2bf(bf2f(xn[i]) * mask);
    }
};
struct EpiStoreF32 { // plain fp32 store
    float* out; int ld;
    __device__ void operator()(int m, int n, float acc) const {
        out[(size_t)m * ld + n] = acc;
    }
};
struct EpiSoftplus { // delta = softplus(acc + bias)  (fp32 out)
    const float* bias; float* out;
    __device__ void operator()(int m, int n, float acc) const {
        float v  = acc + bias[n];
        float sp = (v > 20.f) ? v : log1pf(__expf(v));
        out[(size_t)m * DIM + n] = sp;
    }
};

// ---------------------------------------------------------------- WMMA GEMM
// C(M x N) = A(M x K) * W(N x K)^T.
// Block = 256 thr = 8 waves; block tile 256 x 64; each wave computes a 32 x 64
// strip: 2 A-frags x 4 B-frags = 8 WMMAs per 32-K step (each B reused by 2 A),
// 12 b128 loads per 8 WMMAs. All 8 waves share the same B columns (L0 broadcast).
template <bool AF32, typename EPI>
__global__ __launch_bounds__(256)
void gemm_wmma_kernel(const void* Aptr, int lda,
                      const unsigned short* W, int ldw,
                      int K, EPI epi) {
    const int lane = threadIdx.x & 31;
    const int wave = threadIdx.x >> 5;
    const int m0 = blockIdx.y * 256 + wave * 32;
    const int n0 = blockIdx.x * 64;

    v8f acc[2][4] = {};

    for (int k0 = 0; k0 < K; k0 += 32) {
        v16bf a0, a1;
        if constexpr (AF32) {
            a0 = load_a_f32((const float*)Aptr, lda, m0,      k0, lane);
            a1 = load_a_f32((const float*)Aptr, lda, m0 + 16, k0, lane);
        } else {
            a0 = load_a_bf16((const unsigned short*)Aptr, lda, m0,      k0, lane);
            a1 = load_a_bf16((const unsigned short*)Aptr, lda, m0 + 16, k0, lane);
        }
#pragma unroll
        for (int j = 0; j < 4; ++j) {
            v16bf b = load_b(W, ldw, n0 + 16 * j, k0, lane);
            acc[0][j] = __builtin_amdgcn_wmma_f32_16x16x32_bf16(false, a0, false, b, (short)0, acc[0][j], false, false);
            acc[1][j] = __builtin_amdgcn_wmma_f32_16x16x32_bf16(false, a1, false, b, (short)0, acc[1][j], false, false);
        }
    }

    // C/D layout: VGPR r -> row mfrag + 8*(lane>>4) + r, col nfrag + (lane&15)
    const int nc = lane & 15;
    const int mb = (lane >> 4) * 8;
#pragma unroll
    for (int i = 0; i < 2; ++i)
#pragma unroll
        for (int r = 0; r < 8; ++r)
#pragma unroll
            for (int j = 0; j < 4; ++j)
                epi(m0 + 16 * i + mb + r, n0 + 16 * j + nc, acc[i][j][r]);
}

// ---------------------------------------------------------------- small kernels
__global__ void cvt_bf16_kernel(const float* src, unsigned short* dst, int n) {
    int i = blockIdx.x * blockDim.x + threadIdx.x;
    if (i < n) dst[i] = f2bf(src[i]);
}

__global__ __launch_bounds__(256)
void rmsnorm_kernel(const float* __restrict__ x, const float* __restrict__ w,
                    unsigned short* __restrict__ xn) {
    const int row = blockIdx.x;                 // 0 .. MTOK-1
    const float* xr = x + (size_t)row * DIM;
    float ss = 0.f;
    for (int i = threadIdx.x; i < DIM; i += 256) {
        float v = xr[i];
        ss += v * v;
    }
#pragma unroll
    for (int off = 16; off > 0; off >>= 1) ss += __shfl_xor(ss, off, 32);
    __shared__ float red[8];
    if ((threadIdx.x & 31) == 0) red[threadIdx.x >> 5] = ss;
    __syncthreads();
    if (threadIdx.x == 0) {
        float s = 0.f;
#pragma unroll
        for (int i = 0; i < 8; ++i) s += red[i];
        red[0] = rsqrtf(s / (float)DIM + 1e-6f);
    }
    __syncthreads();
    const float rs = red[0];
    for (int i = threadIdx.x; i < DIM; i += 256)
        xn[(size_t)row * DIM + i] = f2bf(xr[i] * rs * w[i]);
}

// Scan: one wave per (b,d), 4 states per lane (N=128). B/C staged in LDS per block.
// Fuses residual: z = y + x * D_param  (bf16 out for final GEMM).
__global__ __launch_bounds__(256)
void scan_kernel(const float* __restrict__ delta, const float* __restrict__ xp,
                 const unsigned short* __restrict__ xg, const float* __restrict__ x,
                 const float* __restrict__ A_log, const float* __restrict__ Dp,
                 unsigned short* __restrict__ z) {
    __shared__ float sB[NS];
    __shared__ float sC[NS];

    const int b     = blockIdx.x >> 8;           // DIM/8 = 256 blocks per batch
    const int dbase = (blockIdx.x & 255) * 8;
    const int wave  = threadIdx.x >> 5;
    const int lane  = threadIdx.x & 31;
    const int d     = dbase + wave;

    float4 al = *(const float4*)(A_log + (size_t)d * NS + lane * 4);
    const float a0 = -__expf(al.x), a1 = -__expf(al.y);
    const float a2 = -__expf(al.z), a3 = -__expf(al.w);
    const float dparam = Dp[d];

    float h0 = 0.f, h1 = 0.f, h2 = 0.f, h3 = 0.f;

    for (int t = 0; t < SEQ_L; ++t) {
        const size_t row = (size_t)b * SEQ_L + t;
        if (threadIdx.x < NS)
            sB[threadIdx.x] = xp[row * XPLD + RS + threadIdx.x];
        else
            sC[threadIdx.x - NS] = xp[row * XPLD + RS + NS + (threadIdx.x - NS)];
        __syncthreads();

        const float dt = delta[row * DIM + d];
        const float xt = bf2f(xg[row * DIM + d]);
        const float dx = dt * xt;

        float4 bt = *(const float4*)(&sB[lane * 4]);
        float4 ct = *(const float4*)(&sC[lane * 4]);

        h0 = __expf(dt * a0) * h0 + dx * bt.x;
        h1 = __expf(dt * a1) * h1 + dx * bt.y;
        h2 = __expf(dt * a2) * h2 + dx * bt.z;
        h3 = __expf(dt * a3) * h3 + dx * bt.w;

        float y = h0 * ct.x + h1 * ct.y + h2 * ct.z + h3 * ct.w;
#pragma unroll
        for (int off = 16; off > 0; off >>= 1) y += __shfl_xor(y, off, 32);

        if (lane == 0)
            z[row * DIM + d] = f2bf(y + x[row * DIM + d] * dparam);
        __syncthreads();
    }
}

// ---------------------------------------------------------------- launcher
extern "C" void kernel_launch(void* const* d_in, const int* in_sizes, int n_in,
                              void* d_out, int out_size, void* d_ws, size_t ws_size,
                              hipStream_t stream) {
    const float* x      = (const float*)d_in[0];
    const float* norm_w = (const float*)d_in[1];
    const float* sel_w  = (const float*)d_in[2];
    const float* sel_b  = (const float*)d_in[3];
    const float* sel_g  = (const float*)d_in[4];
    const float* xproj_w = (const float*)d_in[5];
    const float* dt_w   = (const float*)d_in[6];
    const float* dt_b   = (const float*)d_in[7];
    const float* A_log  = (const float*)d_in[8];
    const float* D_par  = (const float*)d_in[9];
    const float* out_w  = (const float*)d_in[10];
    float* out = (float*)d_out;

    char* base = (char*)d_ws;
    size_t off = 0;
    auto carve = [&](size_t bytes) -> char* {
        char* p = base + off;
        off += (bytes + 255) & ~(size_t)255;
        return p;
    };
    unsigned short* wsel_bf = (unsigned short*)carve((size_t)DIM * DIM * 2);
    unsigned short* wxp_bf  = (unsigned short*)carve((size_t)XPLD * DIM * 2);
    unsigned short* wdt_bf  = (unsigned short*)carve((size_t)DIM * RS * 2);
    unsigned short* wout_bf = (unsigned short*)carve((size_t)DIM * DIM * 2);
    unsigned short* xn_bf   = (unsigned short*)carve((size_t)MTOK * DIM * 2);
    unsigned short* xg_bf   = (unsigned short*)carve((size_t)MTOK * DIM * 2);
    float*          xp_f    = (float*)carve((size_t)MTOK * XPLD * 4);
    float*          delta_f = (float*)carve((size_t)MTOK * DIM * 4);
    unsigned short* z_bf    = (unsigned short*)carve((size_t)MTOK * DIM * 2);
    (void)ws_size; (void)in_sizes; (void)n_in; (void)out_size;

    // 1) weights -> bf16
    auto cvt = [&](const float* s, unsigned short* d, int n) {
        cvt_bf16_kernel<<<(n + 255) / 256, 256, 0, stream>>>(s, d, n);
    };
    cvt(sel_w,   wsel_bf, DIM * DIM);
    cvt(xproj_w, wxp_bf,  XPLD * DIM);
    cvt(dt_w,    wdt_bf,  DIM * RS);
    cvt(out_w,   wout_bf, DIM * DIM);

    // 2) RMSNorm -> xn (bf16)
    rmsnorm_kernel<<<MTOK, 256, 0, stream>>>(x, norm_w, xn_bf);

    // 3) select GEMM + fused sigmoid gate -> xg (bf16)
    gemm_wmma_kernel<false><<<dim3(DIM / 64, MTOK / 256), 256, 0, stream>>>(
        (const void*)xn_bf, DIM, wsel_bf, DIM, DIM,
        EpiSelect{sel_b, sel_g, xn_bf, xg_bf});

    // 4) x_proj GEMM -> xp (fp32, 4096 x 384)
    gemm_wmma_kernel<false><<<dim3(XPLD / 64, MTOK / 256), 256, 0, stream>>>(
        (const void*)xg_bf, DIM, wxp_bf, DIM, DIM,
        EpiStoreF32{xp_f, XPLD});

    // 5) dt_proj GEMM (A = fp32 slice of xp, K = R = 128) + softplus -> delta (fp32)
    gemm_wmma_kernel<true><<<dim3(DIM / 64, MTOK / 256), 256, 0, stream>>>(
        (const void*)xp_f, XPLD, wdt_bf, RS, RS,
        EpiSoftplus{dt_b, delta_f});

    // 6) selective scan, fused residual -> z (bf16)
    scan_kernel<<<B_SZ * (DIM / 8), 256, 0, stream>>>(
        delta_f, xp_f, xg_bf, x, A_log, D_par, z_bf);

    // 7) out_proj GEMM -> fp32 output
    gemm_wmma_kernel<false><<<dim3(DIM / 64, MTOK / 256), 256, 0, stream>>>(
        (const void*)z_bf, DIM, wout_bf, DIM, DIM,
        EpiStoreF32{out, DIM});
}
